// TripletLossHardMining_90099823936181
// MI455X (gfx1250) — compile-verified, matrix operand order checked by reading, not run on hardware
//
#include <hip/hip_runtime.h>

#define NB 8192
#define ND 128
#define NSPLIT 8                       // N-dimension split for occupancy
#define NCOLS (NB / NSPLIT)            // 1024 columns per split
#define TL (NCOLS / 16)                // 64 tiles per split
#define MARGINF 0.1f
#define BIGF 1e9f

typedef __attribute__((ext_vector_type(16))) __bf16 v16bf;
typedef __attribute__((ext_vector_type(8)))  __bf16 v8bf;
typedef __attribute__((ext_vector_type(4)))  __bf16 v4bf;
typedef __attribute__((ext_vector_type(8)))  float  v8f;

// ---------------------------------------------------------------------------
// Kernel 1: L2-normalize each row (eps 1e-12) and split fp32 -> bf16 hi + lo
// ---------------------------------------------------------------------------
__global__ __launch_bounds__(256) void k_norm_split(
    const float* __restrict__ emb,
    __bf16* __restrict__ xhi,
    __bf16* __restrict__ xlo)
{
  const int wave = threadIdx.x >> 5;
  const int lane = threadIdx.x & 31;
  const int row  = blockIdx.x * 8 + wave;

  float4 v = ((const float4*)(emb + (size_t)row * ND))[lane];
  float ss = v.x * v.x + v.y * v.y + v.z * v.z + v.w * v.w;
#pragma unroll
  for (int m = 1; m < 32; m <<= 1) ss += __shfl_xor(ss, m, 32);

  const float inv = 1.0f / fmaxf(sqrtf(ss), 1e-12f);
  float xs[4] = {v.x * inv, v.y * inv, v.z * inv, v.w * inv};

  v4bf h, l;
#pragma unroll
  for (int i = 0; i < 4; ++i) {
    h[i] = (__bf16)xs[i];
    l[i] = (__bf16)(xs[i] - (float)h[i]);
  }
  *(v4bf*)(xhi + (size_t)row * ND + lane * 4) = h;
  *(v4bf*)(xlo + (size_t)row * ND + lane * 4) = l;
}

// ---------------------------------------------------------------------------
// Kernel 2: fused  dist = 1 - Xn·Xn^T  +  batch-hard mining (partial per split)
// Grid: (64, 8) x 256 threads. blockIdx.x -> 128-row M chunk (wave w owns 16
// rows), blockIdx.y -> 1024-column N split. Double-buffered LDS tile staging,
// bf16x3 compensated WMMA (hi·hi + hi·lo + lo·hi, f32 accumulate).
// Writes per-row partial ap / an_sg / an_all for its split.
// ---------------------------------------------------------------------------
__global__ __launch_bounds__(256) void k_triplet(
    const __bf16* __restrict__ xhi,
    const __bf16* __restrict__ xlo,
    const int*    __restrict__ labels,
    const int*    __restrict__ groups,
    float*        __restrict__ ap_part,
    float*        __restrict__ ansg_part,
    float*        __restrict__ anall_part)
{
  // padded stride 136 (272B rows): b128 LDS reads start at bank 4*n
  __shared__ alignas(16) __bf16 sh_hi[2][16 * 136];
  __shared__ alignas(16) __bf16 sh_lo[2][16 * 136];
  __shared__ int sh_lab[NCOLS];
  __shared__ int sh_grp[NCOLS];

  const int tid   = threadIdx.x;
  const int wave  = tid >> 5;
  const int lane  = tid & 31;
  const int lhalf = lane >> 4;
  const int l15   = lane & 15;
  const int m0    = blockIdx.x * 128 + wave * 16;
  const int split = blockIdx.y;
  const int nbase = split * NCOLS;

  // ---- preload A fragments: 16 rows x K=128 (4 K-steps of 32) ------------
  const int am = m0 + l15;
  const int kb = lhalf * 8;
  v16bf ahi[4], alo[4];
#pragma unroll
  for (int s = 0; s < 4; ++s) {
    const size_t o0 = (size_t)am * ND + 32 * s + kb;
    v8bf h0 = *(const v8bf*)(xhi + o0);
    v8bf h1 = *(const v8bf*)(xhi + o0 + 16);
    v8bf l0 = *(const v8bf*)(xlo + o0);
    v8bf l1 = *(const v8bf*)(xlo + o0 + 16);
#pragma unroll
    for (int i = 0; i < 8; ++i) {
      ahi[s][i] = h0[i]; ahi[s][i + 8] = h1[i];
      alo[s][i] = l0[i]; alo[s][i + 8] = l1[i];
    }
  }

  // ---- per-lane row metadata (C layout: VGPR r -> M = r + 8*lhalf) -------
  int mrow[8], labm[8], grpm[8];
#pragma unroll
  for (int r = 0; r < 8; ++r) {
    mrow[r] = m0 + r + 8 * lhalf;
    labm[r] = labels[mrow[r]];
    grpm[r] = groups[mrow[r]];
  }

  float ap[8], ansg[8], anall[8];
#pragma unroll
  for (int r = 0; r < 8; ++r) { ap[r] = -BIGF; ansg[r] = BIGF; anall[r] = BIGF; }

  // staging indices for this thread
  const int sn = tid >> 4;          // 0..15 : tile row (= column of dist)
  const int sk = (tid & 15) * 8;    // 0..120: chunk of 8 bf16 (16 bytes)

  // ---- prologue: labels/groups for this split + tile 0 into buffer 0 -----
  {
    *(int4*)&sh_lab[tid * 4] = *(const int4*)(labels + nbase + tid * 4);
    *(int4*)&sh_grp[tid * 4] = *(const int4*)(groups + nbase + tid * 4);
    const size_t g = (size_t)(nbase + sn) * ND + sk;
    *(float4*)&sh_hi[0][sn * 136 + sk] = *(const float4*)(xhi + g);
    *(float4*)&sh_lo[0][sn * 136 + sk] = *(const float4*)(xlo + g);
  }
  __syncthreads();

  for (int t = 0; t < TL; ++t) {
    // ---- issue next tile's global loads early (hide latency) -------------
    float4 nh, nl;
    const bool have_next = (t + 1 < TL);
    if (have_next) {
      const size_t g = (size_t)(nbase + (t + 1) * 16 + sn) * ND + sk;
      nh = *(const float4*)(xhi + g);
      nl = *(const float4*)(xlo + g);
    }

    // ---- 16x16 tile GEMM: 4 K-steps x 3 compensated WMMAs ----------------
    const __bf16* bufh = sh_hi[t & 1];
    const __bf16* bufl = sh_lo[t & 1];
    v8f acc = {};
#pragma unroll
    for (int s = 0; s < 4; ++s) {
      const int o = l15 * 136 + 32 * s + kb;   // B layout mirrors A
      v8bf bh0 = *(const v8bf*)(bufh + o);
      v8bf bh1 = *(const v8bf*)(bufh + o + 16);
      v8bf bl0 = *(const v8bf*)(bufl + o);
      v8bf bl1 = *(const v8bf*)(bufl + o + 16);
      v16bf bhi, blo;
#pragma unroll
      for (int i = 0; i < 8; ++i) {
        bhi[i] = bh0[i]; bhi[i + 8] = bh1[i];
        blo[i] = bl0[i]; blo[i + 8] = bl1[i];
      }
      acc = __builtin_amdgcn_wmma_f32_16x16x32_bf16(false, ahi[s], false, bhi,
                                                    (short)0, acc, false, false);
      acc = __builtin_amdgcn_wmma_f32_16x16x32_bf16(false, ahi[s], false, blo,
                                                    (short)0, acc, false, false);
      acc = __builtin_amdgcn_wmma_f32_16x16x32_bf16(false, alo[s], false, bhi,
                                                    (short)0, acc, false, false);
    }

    // ---- masked hard-mining update (metadata from LDS) -------------------
    const int nl16 = t * 16 + l15;        // column index within split
    const int n    = nbase + nl16;        // global column index
    const int labn = sh_lab[nl16];
    const int grpn = sh_grp[nl16];
#pragma unroll
    for (int r = 0; r < 8; ++r) {
      const float d   = 1.0f - acc[r];
      const bool same = (labm[r] == labn);
      const bool pos  = same && (mrow[r] != n);
      const bool neg  = !same;
      const bool sg   = neg && (grpm[r] == grpn);
      if (pos) ap[r]    = fmaxf(ap[r], d);
      if (neg) anall[r] = fminf(anall[r], d);
      if (sg)  ansg[r]  = fminf(ansg[r], d);
    }

    // ---- store next tile into the alternate buffer -----------------------
    if (have_next) {
      const int b = (t + 1) & 1;
      *(float4*)&sh_hi[b][sn * 136 + sk] = nh;
      *(float4*)&sh_lo[b][sn * 136 + sk] = nl;
    }
    __syncthreads();
  }

  // ---- reduce each row's state across its 16 column-lanes ----------------
#pragma unroll
  for (int r = 0; r < 8; ++r) {
#pragma unroll
    for (int m = 1; m < 16; m <<= 1) {
      ap[r]    = fmaxf(ap[r],    __shfl_xor(ap[r],    m, 32));
      ansg[r]  = fminf(ansg[r],  __shfl_xor(ansg[r],  m, 32));
      anall[r] = fminf(anall[r], __shfl_xor(anall[r], m, 32));
    }
  }

  // lanes 0 / 16 hold rows m0+r / m0+8+r -> write per-split partials
  if (l15 == 0) {
#pragma unroll
    for (int r = 0; r < 8; ++r) {
      const int row = mrow[r];
      ap_part[split * NB + row]    = ap[r];
      ansg_part[split * NB + row]  = ansg[r];
      anall_part[split * NB + row] = anall[r];
    }
  }
}

// ---------------------------------------------------------------------------
// Kernel 3: merge the NSPLIT partials per row, compute loss, per-block sums
// ---------------------------------------------------------------------------
__global__ __launch_bounds__(256) void k_combine(
    const float* __restrict__ ap_part,
    const float* __restrict__ ansg_part,
    const float* __restrict__ anall_part,
    float*       __restrict__ blocksums)
{
  __shared__ float sl[8], sc[8];
  const int tid  = threadIdx.x;
  const int wave = tid >> 5;
  const int lane = tid & 31;
  const int row  = blockIdx.x * 256 + tid;

  float ap = -BIGF, ansg = BIGF, anall = BIGF;
#pragma unroll
  for (int s = 0; s < NSPLIT; ++s) {
    ap    = fmaxf(ap,    ap_part[s * NB + row]);
    ansg  = fminf(ansg,  ansg_part[s * NB + row]);
    anall = fminf(anall, anall_part[s * NB + row]);
  }
  // sentinels make loss>0 imply has_pos && has_neg (matches reference)
  const float an   = (ansg < 0.5f * BIGF) ? ansg : anall;
  const float loss = ap - an + MARGINF;
  float ls = (loss > 0.0f) ? loss : 0.0f;
  float lc = (loss > 0.0f) ? 1.0f : 0.0f;
#pragma unroll
  for (int m = 1; m < 32; m <<= 1) {
    ls += __shfl_xor(ls, m, 32);
    lc += __shfl_xor(lc, m, 32);
  }
  if (lane == 0) { sl[wave] = ls; sc[wave] = lc; }
  __syncthreads();
  if (tid == 0) {
    float ts = 0.0f, tc = 0.0f;
#pragma unroll
    for (int w = 0; w < 8; ++w) { ts += sl[w]; tc += sc[w]; }
    blocksums[blockIdx.x * 2 + 0] = ts;
    blocksums[blockIdx.x * 2 + 1] = tc;
  }
}

// ---------------------------------------------------------------------------
// Kernel 4: reduce 32 per-block sums -> scalar loss (deterministic)
// ---------------------------------------------------------------------------
__global__ __launch_bounds__(32) void k_finalize(const float* __restrict__ blocksums,
                                                 float* __restrict__ out)
{
  const int lane = threadIdx.x;
  float ts = blocksums[lane * 2 + 0];
  float tc = blocksums[lane * 2 + 1];
#pragma unroll
  for (int m = 1; m < 32; m <<= 1) {
    ts += __shfl_xor(ts, m, 32);
    tc += __shfl_xor(tc, m, 32);
  }
  if (lane == 0) {
    float f = (tc > 0.0f) ? (ts / fmaxf(tc, 1.0f)) : 0.0f;
    if (f != f) f = 0.0f;  // NaN guard
    out[0] = f;
  }
}

// ---------------------------------------------------------------------------
extern "C" void kernel_launch(void* const* d_in, const int* in_sizes, int n_in,
                              void* d_out, int out_size, void* d_ws, size_t ws_size,
                              hipStream_t stream)
{
  (void)in_sizes; (void)n_in; (void)out_size; (void)ws_size;
  const float* emb    = (const float*)d_in[0];
  const int*   labels = (const int*)d_in[1];
  const int*   groups = (const int*)d_in[2];
  float*       out    = (float*)d_out;

  char* ws = (char*)d_ws;
  const size_t planeB = (size_t)NB * ND * 2;       // 2 MB per bf16 plane
  const size_t partB  = (size_t)NSPLIT * NB * 4;   // 256 KB per partial array
  __bf16* xhi        = (__bf16*)ws;
  __bf16* xlo        = (__bf16*)(ws + planeB);
  float*  ap_part    = (float*)(ws + 2 * planeB);
  float*  ansg_part  = (float*)(ws + 2 * planeB + partB);
  float*  anall_part = (float*)(ws + 2 * planeB + 2 * partB);
  float*  blocksums  = (float*)(ws + 2 * planeB + 3 * partB);

  k_norm_split<<<NB / 8, 256, 0, stream>>>(emb, xhi, xlo);
  dim3 grid(NB / 128, NSPLIT);
  k_triplet<<<grid, 256, 0, stream>>>(xhi, xlo, labels, groups,
                                      ap_part, ansg_part, anall_part);
  k_combine<<<NB / 256, 256, 0, stream>>>(ap_part, ansg_part, anall_part, blocksums);
  k_finalize<<<1, 32, 0, stream>>>(blocksums, out);
}
